// EncoderX_65420941853517
// MI455X (gfx1250) — matrix operand order repaired and checked by reading, not compiled, not run on hardware
//
#include <hip/hip_runtime.h>
#include <hip/hip_bf16.h>

typedef _Float16 v16h __attribute__((ext_vector_type(16)));
typedef float    v8f  __attribute__((ext_vector_type(8)));

#define WMMA(a, b, c) __builtin_amdgcn_wmma_f32_16x16x32_f16(false, (a), false, (b), (short)0, (c), false, false)

__device__ __forceinline__ v8f zero8() {
    v8f z;
#pragma unroll
    for (int i = 0; i < 8; ++i) z[i] = 0.f;
    return z;
}

__device__ __forceinline__ float fast_sigmoid(float x) {
    return 1.f / (1.f + __expf(-x));
}
__device__ __forceinline__ float fast_tanh(float x) {
    float t = __expf(2.f * x);
    return (t - 1.f) / (t + 1.f);
}

__device__ __forceinline__ unsigned pack2h(float a, float b) {
    union { unsigned u; _Float16 h[2]; } t;
    t.h[0] = (_Float16)a; t.h[1] = (_Float16)b;
    return t.u;
}

// K-offset within a 16-bit 16x32 A fragment (ISA 7.12.2 layout)
__device__ __forceinline__ int a_koff(int j, int hi) {
    return (j < 4) ? (2 * j + 8 * hi) : (16 + 2 * (j - 4) + 8 * hi);
}

// Load A fragment from fp16 row-major [rows][K] global memory
__device__ __forceinline__ v16h load_a_f16(const _Float16* base, int row0, int K, int kt, int lane) {
    int m = lane & 15, hi = lane >> 4;
    const unsigned* p = (const unsigned*)base;
    union { v16h v; unsigned u[8]; } r;
#pragma unroll
    for (int j = 0; j < 8; ++j) {
        int k = kt * 32 + a_koff(j, hi);
        r.u[j] = p[((size_t)(row0 + m) * K + k) >> 1];
    }
    return r.v;
}

// Load B fragment from packed-fp16 weight buffer (1 KB per fragment, 32 B per lane)
__device__ __forceinline__ v16h load_b_packed(const unsigned* pw, int frag, int lane) {
    return *(const v16h*)(pw + (size_t)frag * 256 + lane * 8);
}

// ---------------- weight packing: fp32 [N][Ksrc] -> fp16 B fragments ----------------
__global__ __launch_bounds__(256) void pack_b_kernel(const float* __restrict__ W,
                                                     unsigned* __restrict__ dst,
                                                     int Ksrc, int KT, int NT) {
    int e = blockIdx.x * blockDim.x + threadIdx.x;
    int total = KT * NT * 256;
    if (e >= total) return;
    int j = e & 7;
    int lane = (e >> 3) & 31;
    int frag = e >> 8;
    int kt = frag / NT, nt = frag - kt * NT;
    int n = nt * 16 + (lane & 15);
    int k = kt * 32 + a_koff(j, lane >> 4);
    float f0 = (k < Ksrc) ? W[(size_t)n * Ksrc + k] : 0.f;
    float f1 = (k + 1 < Ksrc) ? W[(size_t)n * Ksrc + k + 1] : 0.f;
    dst[(size_t)frag * 256 + lane * 8 + j] = pack2h(f0, f1);
}

// ---------------- LSTM encoder: 8 steps, h kept fp16 in LDS ----------------
#define LSTR 520
__global__ __launch_bounds__(256, 1) void lstm_kernel(
        const float* __restrict__ obs,     // [8,8192,6]
        const float* __restrict__ W_ih,    // [2048,6]
        const float* __restrict__ b_ih,
        const float* __restrict__ b_hh,
        const unsigned* __restrict__ pWhh, // packed fp16 B, KT=16 NT=128
        _Float16* __restrict__ finalh) {   // [8192,512] fp16
    __shared__ _Float16 hbuf[16 * LSTR];
    int rb = blockIdx.x;
    int tid = threadIdx.x, w = tid >> 5, lane = tid & 31;
    int lo = lane & 15, hi = lane >> 4;
    for (int i = tid; i < 16 * LSTR / 2; i += blockDim.x) ((unsigned*)hbuf)[i] = 0u;
    float cst[4][8];
#pragma unroll
    for (int a = 0; a < 4; ++a)
#pragma unroll
        for (int r = 0; r < 8; ++r) cst[a][r] = 0.f;
    __syncthreads();
    const int hb = w * 64;
    for (int t = 0; t < 8; ++t) {
        // recurrent GEMM first, accumulators start at zero; xproj folded in later
        v8f acc[4][4];
#pragma unroll
        for (int q = 0; q < 4; ++q)
#pragma unroll
            for (int j2 = 0; j2 < 4; ++j2) acc[q][j2] = zero8();
#pragma unroll 1
        for (int kt = 0; kt < 16; ++kt) {
            union { v16h v; unsigned u[8]; } a;
#pragma unroll
            for (int j = 0; j < 8; ++j) {
                int k = kt * 32 + a_koff(j, hi);
                a.u[j] = ((const unsigned*)hbuf)[(lo * LSTR + k) >> 1];
            }
#pragma unroll
            for (int q = 0; q < 4; ++q)
#pragma unroll
                for (int j2 = 0; j2 < 4; ++j2) {
                    int nt = q * 32 + w * 4 + j2;
                    v16h b = load_b_packed(pWhh, kt * 128 + nt, lane);
                    acc[q][j2] = WMMA(a.v, b, acc[q][j2]);
                }
        }
        __syncthreads();   // all waves done reading old h
        // add x-projection + biases (short live range: after the WMMA loop)
        float ox[8][6];
#pragma unroll
        for (int r = 0; r < 8; ++r) {
            int row = rb * 16 + r + 8 * hi;
            const float* op = obs + ((size_t)t * 8192 + row) * 6;
#pragma unroll
            for (int d = 0; d < 6; ++d) ox[r][d] = op[d];
        }
#pragma unroll
        for (int q = 0; q < 4; ++q)
#pragma unroll
            for (int j2 = 0; j2 < 4; ++j2) {
                int g = q * 512 + hb + j2 * 16 + lo;
                float base = b_ih[g] + b_hh[g];
                float wv[6];
#pragma unroll
                for (int d = 0; d < 6; ++d) wv[d] = W_ih[g * 6 + d];
#pragma unroll
                for (int r = 0; r < 8; ++r) {
                    float s = base;
#pragma unroll
                    for (int d = 0; d < 6; ++d) s += ox[r][d] * wv[d];
                    acc[q][j2][r] += s;
                }
            }
        // gate math + cell update (all gates for this wave's h-strip are local)
#pragma unroll
        for (int j2 = 0; j2 < 4; ++j2)
#pragma unroll
            for (int r = 0; r < 8; ++r) {
                float iv = fast_sigmoid(acc[0][j2][r]);
                float fv = fast_sigmoid(acc[1][j2][r]);
                float gv = fast_tanh(acc[2][j2][r]);
                float ov = fast_sigmoid(acc[3][j2][r]);
                float c = fv * cst[j2][r] + iv * gv;
                cst[j2][r] = c;
                float h = ov * fast_tanh(c);
                int col = hb + j2 * 16 + lo;
                int mrow = r + 8 * hi;
                hbuf[mrow * LSTR + col] = (_Float16)h;
                if (t == 7)
                    finalh[(size_t)(rb * 16 + mrow) * 512 + col] = (_Float16)h;
            }
        __syncthreads();   // new h visible before next step
    }
}

// ---------------- hM = final_h @ attn_W^T + attn_b -> fp16 ----------------
__global__ __launch_bounds__(256) void gemm_hM(const _Float16* __restrict__ finalh,
                                               const unsigned* __restrict__ pW,
                                               const float* __restrict__ bias,
                                               _Float16* __restrict__ hM) {
    int rb = blockIdx.x, tid = threadIdx.x, w = tid >> 5, lane = tid & 31;
    int lo = lane & 15, hi = lane >> 4;
    v8f acc[2]; acc[0] = zero8(); acc[1] = zero8();
#pragma unroll 1
    for (int kt = 0; kt < 16; ++kt) {
        v16h a = load_a_f16(finalh, rb * 16, 512, kt, lane);
#pragma unroll
        for (int i = 0; i < 2; ++i) {
            v16h b = load_b_packed(pW, kt * 16 + (w * 2 + i), lane);
            acc[i] = WMMA(a, b, acc[i]);
        }
    }
#pragma unroll
    for (int i = 0; i < 2; ++i) {
        int col = (w * 2 + i) * 16 + lo;
        float bv = bias[col];
#pragma unroll
        for (int r = 0; r < 8; ++r) {
            int row = rb * 16 + r + 8 * hi;
            hM[(size_t)row * 256 + col] = (_Float16)(acc[i][r] + bv);
        }
    }
}

// ---------------- per-scene self-attention (P=16 == one WMMA tile) ----------------
__global__ __launch_bounds__(32) void attn_kernel(const _Float16* __restrict__ hM,
                                                  _Float16* __restrict__ ctx) {
    __shared__ _Float16 sA[16 * 18];
    int s = blockIdx.x, lane = threadIdx.x & 31;
    int lo = lane & 15, hi = lane >> 4;
    int r0 = s * 16;
    // scores = hg @ hg^T  (16x16, K=256)
    v8f sc = zero8();
#pragma unroll
    for (int kt = 0; kt < 8; ++kt) {
        v16h a = load_a_f16(hM, r0, 256, kt, lane);
        union { v16h v; unsigned u[8]; } b;  // B[k][n] = hM[r0+n][k]
#pragma unroll
        for (int j = 0; j < 8; ++j) {
            int k = kt * 32 + a_koff(j, hi);
            b.u[j] = ((const unsigned*)hM)[((size_t)(r0 + lo) * 256 + k) >> 1];
        }
        sc = WMMA(a, b.v, sc);
    }
    // row-wise softmax: row elements live across the 16 lanes of each half
#pragma unroll
    for (int r = 0; r < 8; ++r) {
        float x = sc[r];
        float mx = x;
#pragma unroll
        for (int m = 1; m < 16; m <<= 1) mx = fmaxf(mx, __shfl_xor(mx, m, 16));
        float e = __expf(x - mx);
        float sum = e;
#pragma unroll
        for (int m = 1; m < 16; m <<= 1) sum += __shfl_xor(sum, m, 16);
        sA[(r + 8 * hi) * 18 + lo] = (_Float16)(e / sum);
    }
    __syncthreads();
    // A fragment from attn (K=16, upper K zero-padded)
    union { v16h v; unsigned u[8]; } a2;
#pragma unroll
    for (int j = 0; j < 8; ++j) {
        if (hi == 0) {
            int k = 2 * j;
            a2.u[j] = ((const unsigned*)sA)[(lo * 18 + k) >> 1];
        } else a2.u[j] = 0u;
    }
    // ctx = attn @ hg  (16x256)
#pragma unroll 1
    for (int nt = 0; nt < 16; ++nt) {
        union { v16h v; unsigned u[8]; } b;
#pragma unroll
        for (int j = 0; j < 8; ++j) {
            if (hi == 0) {
                int col = nt * 16 + lo;
                b.u[j] = pack2h((float)hM[(size_t)(r0 + 2 * j) * 256 + col],
                                (float)hM[(size_t)(r0 + 2 * j + 1) * 256 + col]);
            } else b.u[j] = 0u;
        }
        v8f c = zero8();
        c = WMMA(a2.v, b.v, c);
#pragma unroll
        for (int r = 0; r < 8; ++r)
            ctx[(size_t)(r0 + r + 8 * hi) * 256 + nt * 16 + lo] = (_Float16)c[r];
    }
}

// ---------------- fc1: [final_h | ctx] -> relu -> hx (fp32 out + fp16 scratch) ----------------
__global__ __launch_bounds__(256) void gemm_fc1(const _Float16* __restrict__ finalh,
                                                const _Float16* __restrict__ ctx,
                                                const unsigned* __restrict__ pW,
                                                const float* __restrict__ bias,
                                                float* __restrict__ out_hx,
                                                _Float16* __restrict__ hx16) {
    int rb = blockIdx.x, tid = threadIdx.x, w = tid >> 5, lane = tid & 31;
    int lo = lane & 15, hi = lane >> 4;
    v8f acc[2]; acc[0] = zero8(); acc[1] = zero8();
#pragma unroll 1
    for (int kt = 0; kt < 24; ++kt) {
        v16h a = (kt < 16) ? load_a_f16(finalh, rb * 16, 512, kt, lane)
                           : load_a_f16(ctx, rb * 16, 256, kt - 16, lane);
#pragma unroll
        for (int i = 0; i < 2; ++i) {
            v16h b = load_b_packed(pW, kt * 16 + (w * 2 + i), lane);
            acc[i] = WMMA(a, b, acc[i]);
        }
    }
#pragma unroll
    for (int i = 0; i < 2; ++i) {
        int col = (w * 2 + i) * 16 + lo;
        float bv = bias[col];
#pragma unroll
        for (int r = 0; r < 8; ++r) {
            int row = rb * 16 + r + 8 * hi;
            float v = fmaxf(acc[i][r] + bv, 0.f);
            out_hx[(size_t)row * 256 + col] = v;
            hx16[(size_t)row * 256 + col] = (_Float16)v;
        }
    }
}

// ---------------- map fc1: [lmf(16384) | lh(9) | pad] @ mfc1_W^T, relu ----------------
// Dominant streaming kernel: 512 MB of fp32 A traffic -> wide b128 loads + prefetch.
__global__ __launch_bounds__(128) void gemm_map1(const float* __restrict__ lmf,
                                                 const float* __restrict__ lh,
                                                 const unsigned* __restrict__ pW,
                                                 const float* __restrict__ bias,
                                                 _Float16* __restrict__ maph) {
    int rb = blockIdx.x, tid = threadIdx.x, w = tid >> 5, lane = tid & 31; // 4 waves
    int lo = lane & 15, hi = lane >> 4;
    const float* arow = lmf + (size_t)(rb * 16 + lo) * 16384;
    v8f acc[4];
#pragma unroll
    for (int i = 0; i < 4; ++i) acc[i] = zero8();
#pragma unroll 1
    for (int kt = 0; kt < 513; ++kt) {
        union { v16h v; unsigned u[8]; } a;
        if (kt < 512) {
            if (kt + 4 < 512) __builtin_prefetch(arow + (kt + 4) * 32, 0, 1);
            // j=0..3 halves cover 8 contiguous floats at kbase, j=4..7 at kbase+16
            int kbase = kt * 32 + 8 * hi;
            float4 f0 = *(const float4*)(arow + kbase);
            float4 f1 = *(const float4*)(arow + kbase + 4);
            float4 f2 = *(const float4*)(arow + kbase + 16);
            float4 f3 = *(const float4*)(arow + kbase + 20);
            a.u[0] = pack2h(f0.x, f0.y);
            a.u[1] = pack2h(f0.z, f0.w);
            a.u[2] = pack2h(f1.x, f1.y);
            a.u[3] = pack2h(f1.z, f1.w);
            a.u[4] = pack2h(f2.x, f2.y);
            a.u[5] = pack2h(f2.z, f2.w);
            a.u[6] = pack2h(f3.x, f3.y);
            a.u[7] = pack2h(f3.z, f3.w);
        } else {
            const float* hrow = lh + (size_t)(rb * 16 + lo) * 9;
#pragma unroll
            for (int j = 0; j < 8; ++j) {
                int k = a_koff(j, hi); // 0..31 within tail tile
                float f0 = (k < 9) ? hrow[k] : 0.f;
                float f1 = (k + 1 < 9) ? hrow[k + 1] : 0.f;
                a.u[j] = pack2h(f0, f1);
            }
        }
#pragma unroll
        for (int i = 0; i < 4; ++i) {
            v16h b = load_b_packed(pW, kt * 16 + (w * 4 + i), lane);
            acc[i] = WMMA(a.v, b, acc[i]);
        }
    }
#pragma unroll
    for (int i = 0; i < 4; ++i) {
        int col = (w * 4 + i) * 16 + lo;
        float bv = bias[col];
#pragma unroll
        for (int r = 0; r < 8; ++r) {
            int row = rb * 16 + r + 8 * hi;
            maph[(size_t)row * 256 + col] = (_Float16)fmaxf(acc[i][r] + bv, 0.f);
        }
    }
}

// ---------------- map fc2: maph @ mfc2_W^T + b -> fp16 ----------------
__global__ __launch_bounds__(256) void gemm_map2(const _Float16* __restrict__ maph,
                                                 const unsigned* __restrict__ pW,
                                                 const float* __restrict__ bias,
                                                 _Float16* __restrict__ mapf) {
    int rb = blockIdx.x, tid = threadIdx.x, w = tid >> 5, lane = tid & 31;
    int lo = lane & 15, hi = lane >> 4;
    v8f acc[2]; acc[0] = zero8(); acc[1] = zero8();
#pragma unroll 1
    for (int kt = 0; kt < 8; ++kt) {
        v16h a = load_a_f16(maph, rb * 16, 256, kt, lane);
#pragma unroll
        for (int i = 0; i < 2; ++i) {
            v16h b = load_b_packed(pW, kt * 16 + (w * 2 + i), lane);
            acc[i] = WMMA(a, b, acc[i]);
        }
    }
#pragma unroll
    for (int i = 0; i < 2; ++i) {
        int col = (w * 2 + i) * 16 + lo;
        float bv = bias[col];
#pragma unroll
        for (int r = 0; r < 8; ++r) {
            int row = rb * 16 + r + 8 * hi;
            mapf[(size_t)row * 256 + col] = (_Float16)(acc[i][r] + bv);
        }
    }
}

// ---------------- fc2: [hx | map_feat] -> mu / logvar (fp32) ----------------
__global__ __launch_bounds__(256) void gemm_fc2(const _Float16* __restrict__ hx16,
                                                const _Float16* __restrict__ mapf,
                                                const unsigned* __restrict__ pW,
                                                const float* __restrict__ bias,
                                                float* __restrict__ outMu,
                                                float* __restrict__ outLv) {
    int rb = blockIdx.x, tid = threadIdx.x, w = tid >> 5, lane = tid & 31;
    int lo = lane & 15, hi = lane >> 4;
    v8f acc[2]; acc[0] = zero8(); acc[1] = zero8();
#pragma unroll 1
    for (int kt = 0; kt < 16; ++kt) {
        v16h a = (kt < 8) ? load_a_f16(hx16, rb * 16, 256, kt, lane)
                          : load_a_f16(mapf, rb * 16, 256, kt - 8, lane);
#pragma unroll
        for (int i = 0; i < 2; ++i) {
            v16h b = load_b_packed(pW, kt * 16 + (w * 2 + i), lane);
            acc[i] = WMMA(a, b, acc[i]);
        }
    }
#pragma unroll
    for (int i = 0; i < 2; ++i) {
        int nt = w * 2 + i;
        int col = nt * 16 + lo;
        float bv = bias[col];
        float* dst = (nt < 8) ? outMu : outLv;
        int dcol = (nt < 8) ? col : (col - 128);
#pragma unroll
        for (int r = 0; r < 8; ++r) {
            int row = rb * 16 + r + 8 * hi;
            dst[(size_t)row * 128 + dcol] = acc[i][r] + bv;
        }
    }
}

extern "C" void kernel_launch(void* const* d_in, const int* in_sizes, int n_in,
                              void* d_out, int out_size, void* d_ws, size_t ws_size,
                              hipStream_t stream) {
    (void)in_sizes; (void)n_in; (void)out_size; (void)ws_size;
    const float* obs    = (const float*)d_in[0];
    const float* lmf    = (const float*)d_in[2];
    const float* lh     = (const float*)d_in[3];
    const float* W_ih   = (const float*)d_in[4];
    const float* W_hh   = (const float*)d_in[5];
    const float* b_ih   = (const float*)d_in[6];
    const float* b_hh   = (const float*)d_in[7];
    const float* attn_W = (const float*)d_in[8];
    const float* attn_b = (const float*)d_in[9];
    const float* fc1_W  = (const float*)d_in[10];
    const float* fc1_b  = (const float*)d_in[11];
    const float* fc2_W  = (const float*)d_in[12];
    const float* fc2_b  = (const float*)d_in[13];
    const float* mfc1_W = (const float*)d_in[14];
    const float* mfc1_b = (const float*)d_in[15];
    const float* mfc2_W = (const float*)d_in[16];
    const float* mfc2_b = (const float*)d_in[17];

    char* ws = (char*)d_ws;
    size_t off = 0;
    auto carve = [&](size_t bytes) -> void* {
        void* p = ws + off;
        off = (off + bytes + 255) & ~(size_t)255;
        return p;
    };
    unsigned*  pWhh   = (unsigned*)carve((size_t)16 * 128 * 1024);
    unsigned*  pAttn  = (unsigned*)carve((size_t)16 * 16 * 1024);
    unsigned*  pFc1   = (unsigned*)carve((size_t)24 * 16 * 1024);
    unsigned*  pFc2   = (unsigned*)carve((size_t)16 * 16 * 1024);
    unsigned*  pM1    = (unsigned*)carve((size_t)513 * 16 * 1024);
    unsigned*  pM2    = (unsigned*)carve((size_t)8 * 16 * 1024);
    _Float16*  finalh = (_Float16*)carve((size_t)8192 * 512 * 2);
    _Float16*  hM     = (_Float16*)carve((size_t)8192 * 256 * 2);
    _Float16*  ctx    = (_Float16*)carve((size_t)8192 * 256 * 2);
    _Float16*  hx16   = (_Float16*)carve((size_t)8192 * 256 * 2);
    _Float16*  maph   = (_Float16*)carve((size_t)8192 * 256 * 2);
    _Float16*  mapf   = (_Float16*)carve((size_t)8192 * 256 * 2);

    auto pack = [&](const float* W, unsigned* dst, int Ksrc, int KT, int NT) {
        int total = KT * NT * 256;
        pack_b_kernel<<<(total + 255) / 256, 256, 0, stream>>>(W, dst, Ksrc, KT, NT);
    };
    pack(W_hh,   pWhh,  512,   16,  128);
    pack(attn_W, pAttn, 512,   16,  16);
    pack(fc1_W,  pFc1,  768,   24,  16);
    pack(fc2_W,  pFc2,  512,   16,  16);
    pack(mfc1_W, pM1,   16393, 513, 16);
    pack(mfc2_W, pM2,   256,   8,   16);

    lstm_kernel<<<512, 256, 0, stream>>>(obs, W_ih, b_ih, b_hh, pWhh, finalh);
    gemm_hM<<<512, 256, 0, stream>>>(finalh, pAttn, attn_b, hM);
    attn_kernel<<<512, 32, 0, stream>>>(hM, ctx);

    float* out_hx = (float*)d_out;
    gemm_fc1<<<512, 256, 0, stream>>>(finalh, ctx, pFc1, fc1_b, out_hx, hx16);

    gemm_map1<<<512, 128, 0, stream>>>(lmf, lh, pM1, mfc1_b, maph);
    gemm_map2<<<512, 256, 0, stream>>>(maph, pM2, mfc2_b, mapf);

    float* outMu = out_hx + (size_t)8192 * 256;
    float* outLv = outMu + (size_t)8192 * 128;
    gemm_fc2<<<512, 256, 0, stream>>>(hx16, mapf, pFc2, fc2_b, outMu, outLv);
}
